// RegionProposalNetwork_72602127172321
// MI455X (gfx1250) — matrix operand order repaired and checked by reading, not compiled, not run on hardware
//
#include <hip/hip_runtime.h>

// ---------------------------------------------------------------------------
// Types for CDNA5 WMMA (gfx1250, wave32)
// ---------------------------------------------------------------------------
typedef __attribute__((ext_vector_type(16))) __bf16    v16bf;
typedef __attribute__((ext_vector_type(8)))  float     v8f;
typedef __attribute__((ext_vector_type(4)))  unsigned  v4u;

union Frag {            // 8 VGPRs = 16 bf16 elements per lane
  v16bf bf;
  v4u   q[2];
};

// fp32 -> bf16 round-to-nearest-even, returned as raw bits
__device__ __forceinline__ unsigned short f2bf(float f) {
  unsigned u = __float_as_uint(f);
  unsigned r = (u + 0x7FFFu + ((u >> 16) & 1u)) >> 16;
  return (unsigned short)r;
}

// float -> totally-ordered uint key (ascending key == ascending float)
__device__ __forceinline__ unsigned orderKey(float f) {
  unsigned u = __float_as_uint(f);
  return (u & 0x80000000u) ? ~u : (u | 0x80000000u);
}

// Problem constants
#define NB   8        // batch
#define HH   64
#define WW   64
#define CC   256      // channels in/out of conv1
#define NPIX 4096     // H*W
#define MTOT 32768    // NB*NPIX
#define KTOT 2304     // 9*CC
#define NAA  9
#define NANCH 36864   // NPIX*NAA
#define NPRE 6000
#define NPOST 300
#define PH   66       // padded height (halo = 1)
#define PW   66       // padded width
#define XPTOT ((size_t)NB * PH * PW * CC)   // padded bf16 input elements

// ---------------------------------------------------------------------------
// Pack kernels: fp32 -> bf16 layouts for WMMA
// ---------------------------------------------------------------------------
__global__ void zero_pad_kernel(unsigned short* __restrict__ xp) {
  size_t gid = (size_t)blockIdx.x * blockDim.x + threadIdx.x;
  if (gid < XPTOT) xp[gid] = 0;
}

__global__ void pack_x_kernel(const float* __restrict__ x,     // [8][256][64][64]
                              unsigned short* __restrict__ xp) // [8][66][66][256] padded NHWC
{
  int gid = blockIdx.x * blockDim.x + threadIdx.x;     // over input index
  if (gid >= NB * CC * NPIX) return;
  int pix = gid & (NPIX - 1);
  int c   = (gid >> 12) & (CC - 1);
  int n   = gid >> 20;
  int y   = pix >> 6, xx = pix & 63;
  xp[(((size_t)n * PH + (y + 1)) * PW + (xx + 1)) * CC + c] = f2bf(x[gid]);
}

__global__ void pack_w1_kernel(const float* __restrict__ w,     // [256][256][3][3] OIHW
                               unsigned short* __restrict__ wb) // [256][2304], k=(kh*3+kw)*256+cin
{
  int gid = blockIdx.x * blockDim.x + threadIdx.x;
  if (gid >= CC * KTOT) return;
  int k    = gid % KTOT;
  int cout = gid / KTOT;
  int khw  = k >> 8;            // 0..8
  int cin  = k & 255;
  wb[gid] = f2bf(w[(cout * CC + cin) * 9 + khw]);
}

__global__ void pack_wsl_kernel(const float* __restrict__ locw,   // [36][256]
                                const float* __restrict__ scorew, // [18][256]
                                const float* __restrict__ locb,
                                const float* __restrict__ scoreb,
                                unsigned short* __restrict__ wsl,  // [64][256]
                                float* __restrict__ bsl)           // [64]
{
  int gid = blockIdx.x * blockDim.x + threadIdx.x;
  if (gid >= 64 * CC) return;
  int cin = gid & 255;
  int j   = gid >> 8;
  float v = 0.f;
  if (j < 36)      v = locw[j * CC + cin];
  else if (j < 54) v = scorew[(j - 36) * CC + cin];
  wsl[gid] = f2bf(v);
  if (gid < 64) {
    float b = 0.f;
    if (gid < 36)      b = locb[gid];
    else if (gid < 54) b = scoreb[gid - 36];
    bsl[gid] = b;
  }
}

// ---------------------------------------------------------------------------
// 3x3 conv as implicit GEMM on WMMA bf16, zero-padded input (no bounds checks).
// One wave computes a 32(pixel) x 32(cout) tile: 2 A frags x 2 B frags ->
// 4 back-to-back v_wmma per K-chunk (2 b128 loads per WMMA). Register
// double-buffering: chunk i+1 loads are issued before chunk i's WMMAs so the
// matrix pipe overlaps VMEM. 8 waves/block cover all 256 output channels.
// ---------------------------------------------------------------------------
__global__ __launch_bounds__(256) void conv3x3_wmma_kernel(
    const unsigned short* __restrict__ xp,   // [8][66][66][256] padded NHWC bf16
    const unsigned short* __restrict__ wb,   // [256][2304] bf16
    const float* __restrict__ bias,          // [256]
    unsigned short* __restrict__ hout)       // [8][4096][256] NHWC bf16 (ReLU'd)
{
  const int lane = threadIdx.x & 31;
  const int wave = threadIdx.x >> 5;
  const int m0   = blockIdx.x * 32;                    // pixel tile base (32 px)
  const int n    = m0 >> 12;
  const int y    = (m0 >> 6) & 63;
  const int x0   = m0 & 63;                            // 0 or 32
  const int col  = lane & 15;
  const bool hi  = lane >= 16;
  const int cout0 = wave * 32 + col;                   // cout tile 0
  const int cout1 = cout0 + 16;                        // cout tile 1
  const int aoff = hi ? 8 : 0;                         // A K-pair lane offset
  const int boff = hi ? 8 : 0;                         // B K lane offset

  v8f acc00 = {}, acc01 = {}, acc10 = {}, acc11 = {};
  const unsigned short* brow0 = wb + (size_t)cout0 * KTOT + boff;
  const unsigned short* brow1 = wb + (size_t)cout1 * KTOT + boff;
  // A pixel pointer: padded row y+kh, padded col (x0|x0+16)+col+kw
  const unsigned short* abase =
      xp + (((size_t)n * PH + y) * PW + (x0 + col)) * CC + aoff;

  Frag a0[2], a1[2], b0[2], b1[2];

  // prologue: load chunk 0 into slot 0
  {
    a0[0].q[0] = *(const v4u*)(abase);
    a0[0].q[1] = *(const v4u*)(abase + 16);
    a1[0].q[0] = *(const v4u*)(abase + 16 * CC);
    a1[0].q[1] = *(const v4u*)(abase + 16 * CC + 16);
    b0[0].q[0] = *(const v4u*)(brow0);
    b0[0].q[1] = *(const v4u*)(brow0 + 16);
    b1[0].q[0] = *(const v4u*)(brow1);
    b1[0].q[1] = *(const v4u*)(brow1 + 16);
  }

  #pragma unroll
  for (int kc = 0; kc < 72; ++kc) {       // chunk = (tap 0..8) x (cin/32 0..7)
    const int cur = kc & 1;
    const int nxt = cur ^ 1;
    if (kc + 1 < 72) {                     // prefetch next chunk into other slot
      const int kn  = kc + 1;
      const int tap = kn >> 3;
      const int c0  = (kn & 7) << 5;
      const int kh  = tap / 3;
      const int kw  = tap - 3 * kh;
      const unsigned short* ap = abase + (kh * PW + kw) * CC + c0;
      const unsigned short* p0 = brow0 + tap * 256 + c0;
      const unsigned short* p1 = brow1 + tap * 256 + c0;
      a0[nxt].q[0] = *(const v4u*)(ap);
      a0[nxt].q[1] = *(const v4u*)(ap + 16);
      a1[nxt].q[0] = *(const v4u*)(ap + 16 * CC);
      a1[nxt].q[1] = *(const v4u*)(ap + 16 * CC + 16);
      b0[nxt].q[0] = *(const v4u*)(p0);
      b0[nxt].q[1] = *(const v4u*)(p0 + 16);
      b1[nxt].q[0] = *(const v4u*)(p1);
      b1[nxt].q[1] = *(const v4u*)(p1 + 16);
    }
    acc00 = __builtin_amdgcn_wmma_f32_16x16x32_bf16(
        false, a0[cur].bf, false, b0[cur].bf, (short)0, acc00, false, false);
    acc01 = __builtin_amdgcn_wmma_f32_16x16x32_bf16(
        false, a0[cur].bf, false, b1[cur].bf, (short)0, acc01, false, false);
    acc10 = __builtin_amdgcn_wmma_f32_16x16x32_bf16(
        false, a1[cur].bf, false, b0[cur].bf, (short)0, acc10, false, false);
    acc11 = __builtin_amdgcn_wmma_f32_16x16x32_bf16(
        false, a1[cur].bf, false, b1[cur].bf, (short)0, acc11, false, false);
  }

  const float bv0 = bias[cout0], bv1 = bias[cout1];
  const int mb0 = m0 + (hi ? 8 : 0);        // A-tile 0 rows
  const int mb1 = mb0 + 16;                 // A-tile 1 rows
  #pragma unroll
  for (int r = 0; r < 8; ++r) {
    float v00 = acc00[r] + bv0, v01 = acc01[r] + bv1;
    float v10 = acc10[r] + bv0, v11 = acc11[r] + bv1;
    v00 = v00 > 0.f ? v00 : 0.f;
    v01 = v01 > 0.f ? v01 : 0.f;
    v10 = v10 > 0.f ? v10 : 0.f;
    v11 = v11 > 0.f ? v11 : 0.f;
    hout[(size_t)(mb0 + r) * CC + cout0] = f2bf(v00);
    hout[(size_t)(mb0 + r) * CC + cout1] = f2bf(v01);
    hout[(size_t)(mb1 + r) * CC + cout0] = f2bf(v10);
    hout[(size_t)(mb1 + r) * CC + cout1] = f2bf(v11);
  }
}

// ---------------------------------------------------------------------------
// 1x1 convs (loc 36 + score 18, padded to 64) as WMMA GEMM: M=32768,K=256,N=64
// ---------------------------------------------------------------------------
__global__ __launch_bounds__(128) void gemm64_wmma_kernel(
    const unsigned short* __restrict__ hbf,  // [32768][256] bf16
    const unsigned short* __restrict__ wsl,  // [64][256] bf16
    const float* __restrict__ bsl,           // [64]
    float* __restrict__ out)                 // [32768][64]
{
  const int lane = threadIdx.x & 31;
  const int ct   = threadIdx.x >> 5;         // 0..3 : cout tile
  const int m0   = blockIdx.x * 16;
  const int col  = lane & 15;
  const bool hi  = lane >= 16;
  const int cout = ct * 16 + col;

  v8f acc = {};
  const unsigned short* brow = wsl + cout * CC + (hi ? 8 : 0);
  const unsigned short* ap   = hbf + (size_t)(m0 + col) * CC + (hi ? 8 : 0);
  #pragma unroll
  for (int c0 = 0; c0 < CC; c0 += 32) {
    Frag a, b;
    a.q[0] = *(const v4u*)(ap + c0);
    a.q[1] = *(const v4u*)(ap + c0 + 16);
    b.q[0] = *(const v4u*)(brow + c0);
    b.q[1] = *(const v4u*)(brow + c0 + 16);
    acc = __builtin_amdgcn_wmma_f32_16x16x32_bf16(
        false, a.bf, false, b.bf, (short)0, acc, false, false);
  }
  const float bv = bsl[cout];
  const int mbase = m0 + (hi ? 8 : 0);
  #pragma unroll
  for (int r = 0; r < 8; ++r)
    out[(size_t)(mbase + r) * 64 + cout] = acc[r] + bv;
}

// ---------------------------------------------------------------------------
// Fused postprocess: emit rpn_loc/rpn_score/anchors, softmax fg, decode+clip,
// min-size filter, orderable score keys.
// ---------------------------------------------------------------------------
__global__ void postprocess_kernel(
    const float* __restrict__ sl,        // [32768][64]
    const int* __restrict__ imh, const int* __restrict__ imw,
    float* __restrict__ outLoc,          // [8][36864][4]
    float* __restrict__ outScore,        // [8][36864][2]
    float* __restrict__ outAnch,         // [36864][4]
    float* __restrict__ boxes,           // [8][36864][4]
    unsigned* __restrict__ keys)         // [8][36864]
{
  int gid = blockIdx.x * blockDim.x + threadIdx.x;
  if (gid >= NB * NPIX * NAA) return;
  int a   = gid % NAA;
  int pix = (gid / NAA) & (NPIX - 1);
  int n   = gid / (NAA * NPIX);
  int m   = n * NPIX + pix;
  const float* row = sl + (size_t)m * 64;

  float dy = row[a * 4 + 0], dx = row[a * 4 + 1];
  float dh = row[a * 4 + 2], dw = row[a * 4 + 3];
  int li = n * NANCH + pix * NAA + a;
  outLoc[li * 4 + 0] = dy; outLoc[li * 4 + 1] = dx;
  outLoc[li * 4 + 2] = dh; outLoc[li * 4 + 3] = dw;

  float s0 = row[36 + a * 2], s1 = row[36 + a * 2 + 1];
  outScore[li * 2 + 0] = s0; outScore[li * 2 + 1] = s1;
  float fg = 1.f / (1.f + __expf(s0 - s1));   // softmax [...,1]

  // anchor for (pix, a)
  const float ratios[3] = {0.5f, 1.0f, 2.0f};
  const float scales[3] = {8.0f, 16.0f, 32.0f};
  float r = ratios[a / 3], s = scales[a % 3];
  float ahl = 7.0f * s * __fsqrt_rn(r);
  float awl = 7.0f * s * __fsqrt_rn(1.0f / r);
  int yy = pix >> 6, xx = pix & 63;
  float sy = yy * 16.0f, sx = xx * 16.0f;
  float a0 = sy + 3.5f - 0.5f * ahl;
  float a1 = sx + 3.5f - 0.5f * awl;
  float a2 = sy + 3.5f + 0.5f * ahl;
  float a3 = sx + 3.5f + 0.5f * awl;
  if (n == 0) {
    int ai = pix * NAA + a;
    outAnch[ai * 4 + 0] = a0; outAnch[ai * 4 + 1] = a1;
    outAnch[ai * 4 + 2] = a2; outAnch[ai * 4 + 3] = a3;
  }

  // decode + clip
  float ah = a2 - a0, aw = a3 - a1;
  float acy = a0 + 0.5f * ah, acx = a1 + 0.5f * aw;
  float cy = dy * ah + acy, cx = dx * aw + acx;
  float bh = __expf(dh) * ah, bw = __expf(dw) * aw;
  float fh = (float)imh[0], fw = (float)imw[0];
  float b0 = fminf(fmaxf(cy - 0.5f * bh, 0.f), fh);
  float b1 = fminf(fmaxf(cx - 0.5f * bw, 0.f), fw);
  float b2 = fminf(fmaxf(cy + 0.5f * bh, 0.f), fh);
  float b3 = fminf(fmaxf(cx + 0.5f * bw, 0.f), fw);
  boxes[li * 4 + 0] = b0; boxes[li * 4 + 1] = b1;
  boxes[li * 4 + 2] = b2; boxes[li * 4 + 3] = b3;

  bool valid = (b2 - b0 >= 16.0f) && (b3 - b1 >= 16.0f);
  keys[li] = orderKey(valid ? fg : -__builtin_huge_valf());
}

// ---------------------------------------------------------------------------
// Per-image top-6000: LDS 2048-bin histogram select + 8192-entry LDS bitonic
// sort (~74 KB LDS; WGP has 320 KB). One workgroup per image.
// ---------------------------------------------------------------------------
__global__ __launch_bounds__(1024) void topk_kernel(
    const unsigned* __restrict__ keysG,     // [8][36864]
    const float* __restrict__ boxes,        // [8][36864][4]
    float* __restrict__ boxesK)             // [8][6000][4] sorted by score desc
{
  extern __shared__ unsigned smem_u[];
  unsigned* skey = smem_u;                  // 8192
  unsigned* sidx = smem_u + 8192;           // 8192
  unsigned* hist = smem_u + 16384;          // 2048
  unsigned* ctrl = smem_u + 18432;          // 4

  const int n = blockIdx.x, tid = threadIdx.x;
  const unsigned* kg = keysG + n * NANCH;

  for (int b = tid; b < 2048; b += 1024) hist[b] = 0;
  if (tid < 4) ctrl[tid] = 0;
  __syncthreads();
  for (int i = tid; i < NANCH; i += 1024) atomicAdd(&hist[kg[i] >> 21], 1u);
  __syncthreads();
  if (tid == 0) {
    unsigned acc = 0; int B = 0;
    for (int b = 2047; b >= 0; --b) { acc += hist[b]; if (acc >= NPRE) { B = b; break; } }
    ctrl[0] = (unsigned)B;
    ctrl[1] = NPRE - (acc - hist[B]);       // how many to take from boundary bin
  }
  __syncthreads();
  const unsigned B = ctrl[0], takeB = ctrl[1];
  for (int i = tid; i < NANCH; i += 1024) {
    unsigned k = kg[i], b = k >> 21;
    if (b > B) {
      unsigned p = atomicAdd(&ctrl[2], 1u);
      skey[p] = k; sidx[p] = (unsigned)i;
    } else if (b == B) {
      unsigned q = atomicAdd(&ctrl[3], 1u);
      if (q < takeB) {
        unsigned p = atomicAdd(&ctrl[2], 1u);
        skey[p] = k; sidx[p] = (unsigned)i;
      }
    }
  }
  __syncthreads();
  for (int i = NPRE + tid; i < 8192; i += 1024) { skey[i] = 0; sidx[i] = 0; }
  __syncthreads();

  // bitonic sort, descending by key
  for (unsigned k = 2; k <= 8192; k <<= 1) {
    for (unsigned j = k >> 1; j > 0; j >>= 1) {
      for (unsigned i = tid; i < 8192; i += 1024) {
        unsigned l = i ^ j;
        if (l > i) {
          unsigned ki = skey[i], kl = skey[l];
          bool desc = ((i & k) == 0);
          bool sw = desc ? (ki < kl) : (ki > kl);
          if (sw) {
            skey[i] = kl; skey[l] = ki;
            unsigned t = sidx[i]; sidx[i] = sidx[l]; sidx[l] = t;
          }
        }
      }
      __syncthreads();
    }
  }

  const float* bsrc = boxes + (size_t)n * NANCH * 4;
  for (int i = tid; i < NPRE; i += 1024) {
    unsigned id = sidx[i];
    float* dst = boxesK + ((size_t)n * NPRE + i) * 4;
    dst[0] = bsrc[id * 4 + 0]; dst[1] = bsrc[id * 4 + 1];
    dst[2] = bsrc[id * 4 + 2]; dst[3] = bsrc[id * 4 + 3];
  }
}

// ---------------------------------------------------------------------------
// Per-image greedy NMS with all 6000 boxes resident in LDS (~101 KB).
// Emits 300 rois (+ padding with suppressed boxes in order) and roi indices.
// ---------------------------------------------------------------------------
__global__ __launch_bounds__(1024) void nms_kernel(
    const float* __restrict__ boxesK,   // [8][6000][4]
    float* __restrict__ rois,           // [2400][4]
    float* __restrict__ roiIdx)         // [2400]
{
  extern __shared__ unsigned char smem_c[];
  float*         bx  = (float*)smem_c;                       // 6000*4 floats
  unsigned char* sup = smem_c + NPRE * 16;                   // 6000 bytes
  int*           sel = (int*)(smem_c + NPRE * 16 + NPRE);    // 300 ints (4-aligned)

  const int n = blockIdx.x, tid = threadIdx.x;
  for (int i = tid; i < NPRE * 4; i += blockDim.x)
    bx[i] = boxesK[(size_t)n * NPRE * 4 + i];
  for (int i = tid; i < NPRE; i += blockDim.x) sup[i] = 0;
  __syncthreads();

  for (int i = 0; i < NPRE; ++i) {
    if (!sup[i]) {
      float y1 = bx[i * 4], x1 = bx[i * 4 + 1];
      float y2 = bx[i * 4 + 2], x2 = bx[i * 4 + 3];
      float ar = (y2 - y1) * (x2 - x1);
      for (int j = i + 1 + tid; j < NPRE; j += blockDim.x) {
        if (sup[j]) continue;
        float jy1 = bx[j * 4], jx1 = bx[j * 4 + 1];
        float jy2 = bx[j * 4 + 2], jx2 = bx[j * 4 + 3];
        float ih = fminf(y2, jy2) - fmaxf(y1, jy1);
        float iw = fminf(x2, jx2) - fmaxf(x1, jx1);
        float inter = fmaxf(ih, 0.f) * fmaxf(iw, 0.f);
        float arj = (jy2 - jy1) * (jx2 - jx1);
        float iou = inter / (ar + arj - inter + 1e-9f);
        if (iou > 0.7f) sup[j] = 1;
      }
    }
    __syncthreads();
  }

  if (tid == 0) {
    int c = 0;
    for (int i = 0; i < NPRE && c < NPOST; ++i) if (!sup[i]) sel[c++] = i;
    for (int i = 0; i < NPRE && c < NPOST; ++i) if (sup[i])  sel[c++] = i;
  }
  __syncthreads();
  for (int r = tid; r < NPOST; r += blockDim.x) {
    int s = sel[r];
    float* dst = rois + ((size_t)n * NPOST + r) * 4;
    dst[0] = bx[s * 4 + 0]; dst[1] = bx[s * 4 + 1];
    dst[2] = bx[s * 4 + 2]; dst[3] = bx[s * 4 + 3];
    roiIdx[n * NPOST + r] = (float)n;
  }
}

// ---------------------------------------------------------------------------
// Host-side orchestration
// ---------------------------------------------------------------------------
extern "C" void kernel_launch(void* const* d_in, const int* in_sizes, int n_in,
                              void* d_out, int out_size, void* d_ws, size_t ws_size,
                              hipStream_t stream) {
  (void)in_sizes; (void)n_in; (void)out_size; (void)ws_size;
  const float* x   = (const float*)d_in[0];
  const float* w1  = (const float*)d_in[1];
  const float* b1  = (const float*)d_in[2];
  const float* sw  = (const float*)d_in[3];
  const float* sb  = (const float*)d_in[4];
  const float* lw  = (const float*)d_in[5];
  const float* lb  = (const float*)d_in[6];
  const int*   imh = (const int*)d_in[7];
  const int*   imw = (const int*)d_in[8];

  // Workspace partition
  unsigned char* ws = (unsigned char*)d_ws;
  unsigned short* xp  = (unsigned short*)ws;                       // padded NHWC input
  unsigned short* hbf = xp + XPTOT;                                // 8*4096*256
  unsigned short* wb  = hbf + (size_t)MTOT * CC;                   // 256*2304
  unsigned short* wsl = wb + (size_t)CC * KTOT;                    // 64*256
  float* bsl    = (float*)(wsl + 64 * CC);                         // 64
  float* sl     = bsl + 64;                                        // 32768*64
  float* boxes  = sl + (size_t)MTOT * 64;                          // 8*36864*4
  unsigned* keysG = (unsigned*)(boxes + (size_t)NB * NANCH * 4);   // 8*36864
  float* boxesK = (float*)(keysG + (size_t)NB * NANCH);            // 8*6000*4

  // Output partition (floats, concatenated in reference return order)
  float* out       = (float*)d_out;
  float* outLoc    = out;                                // 1,179,648
  float* outScore  = out + (size_t)NB * NANCH * 4;       //   589,824
  float* outRois   = outScore + (size_t)NB * NANCH * 2;  //     9,600
  float* outRoiIdx = outRois + (size_t)NB * NPOST * 4;   //     2,400
  float* outAnch   = outRoiIdx + (size_t)NB * NPOST;     //   147,456

  zero_pad_kernel<<<(int)((XPTOT + 255) / 256), 256, 0, stream>>>(xp);
  pack_x_kernel<<<(NB * CC * NPIX) / 256, 256, 0, stream>>>(x, xp);
  pack_w1_kernel<<<(CC * KTOT) / 256, 256, 0, stream>>>(w1, wb);
  pack_wsl_kernel<<<(64 * CC) / 256, 256, 0, stream>>>(lw, sw, lb, sb, wsl, bsl);

  conv3x3_wmma_kernel<<<MTOT / 32, 256, 0, stream>>>(xp, wb, b1, hbf);
  gemm64_wmma_kernel<<<MTOT / 16, 128, 0, stream>>>(hbf, wsl, bsl, sl);

  postprocess_kernel<<<(NB * NPIX * NAA) / 256, 256, 0, stream>>>(
      sl, imh, imw, outLoc, outScore, outAnch, boxes, keysG);

  size_t topkLds = (8192 + 8192 + 2048 + 4) * sizeof(unsigned);   // ~73.7 KB
  topk_kernel<<<NB, 1024, topkLds, stream>>>(keysG, boxes, boxesK);

  size_t nmsLds = (size_t)NPRE * 16 + NPRE + NPOST * 4 + 16;      // ~103.3 KB
  nms_kernel<<<NB, 1024, nmsLds, stream>>>(boxesK, outRois, outRoiIdx);
}
